// NeuralODEModel_23081154249516
// MI455X (gfx1250) — matrix operand order repaired and checked by reading, not compiled, not run on hardware
//
#include <hip/hip_runtime.h>

#define HIDDEN 256
#define SDIM   3
#define ROWS   16
#define HSTR   258   // padded LDS row stride (bf16 elements): 258*2B = 516B -> conflict-free column reads

typedef __attribute__((ext_vector_type(16))) __bf16       v16bf;
typedef __attribute__((ext_vector_type(8)))  float        v8f;
typedef __attribute__((ext_vector_type(8)))  unsigned int v8u;

static __device__ __forceinline__ unsigned short f2bf(float f) {
  unsigned int u = __builtin_bit_cast(unsigned int, f);
  u += 0x7fffu + ((u >> 16) & 1u);   // round-to-nearest-even
  return (unsigned short)(u >> 16);
}
static __device__ __forceinline__ unsigned int packbf(float lo, float hi) {
  return (unsigned int)f2bf(lo) | ((unsigned int)f2bf(hi) << 16);
}

#if defined(__has_builtin)
#if __has_builtin(__builtin_amdgcn_tanhf)
#define FAST_TANH(x) __builtin_amdgcn_tanhf(x)
#endif
#endif
#ifndef FAST_TANH
#define FAST_TANH(x) tanhf(x)
#endif

__global__ __launch_bounds__(256)
void node_rk4_wmma(const float* __restrict__ x0, const float* __restrict__ tg,
                   const float* __restrict__ W1, const float* __restrict__ b1,
                   const float* __restrict__ W2, const float* __restrict__ b2,
                   const float* __restrict__ W3, const float* __restrict__ b3,
                   float* __restrict__ out, int T)
{
  __shared__ __align__(16) unsigned short sH1[ROWS * HSTR]; // layer-1 activations (bf16)
  __shared__ __align__(16) unsigned short sH2[ROWS * HSTR]; // layer-2 activations (bf16)
  __shared__ float sS[ROWS * 4];    // current state  [16][3] (padded)
  __shared__ float sSe[ROWS * 4];   // vf evaluation state
  __shared__ float sK[ROWS * 4];    // RK4 weighted-k accumulator
  __shared__ float sOut[ROWS * 4];  // vf output (atomic reduction target)

  const int tid  = threadIdx.x;
  const int wid  = tid >> 5;     // wave id, 0..7
  const int lane = tid & 31;
  const int lm   = lane & 15;    // lane within half-wave
  const int half = lane >> 4;    // 0 / 1
  const int b0   = blockIdx.x * ROWS;

  // ---------------- invariant preloads (registers only) ----------------
  // Layer-1 weights: thread == hidden column
  const float w1r0 = W1[0 * HIDDEN + tid];
  const float w1r1 = W1[1 * HIDDEN + tid];
  const float w1r2 = W1[2 * HIDDEN + tid];
  const float b1r  = b1[tid];
  // Layer-2 bias for this wave's two N-tiles
  const float b2r0 = b2[wid * 32 + lm];
  const float b2r1 = b2[wid * 32 + 16 + lm];

  // W2 B-fragments: wave owns N columns [32w, 32w+32).  32x16 bf16 B in contiguous k order (same as A):
  //  lanes 0-15: N=lane, VGPR j holds K = 2j,2j+1 (K 0..15); lanes 16-31: K += 16.
  v16bf Bf0[8], Bf1[8];
  {
    const int n0 = wid * 32 + lm;
#pragma unroll
    for (int kc = 0; kc < 8; ++kc) {
      v8u t0, t1;
#pragma unroll
      for (int j = 0; j < 8; ++j) {
        const int k = kc * 32 + half * 16 + 2 * j;
        t0[j] = packbf(W2[k * HIDDEN + n0],      W2[(k + 1) * HIDDEN + n0]);
        t1[j] = packbf(W2[k * HIDDEN + n0 + 16], W2[(k + 1) * HIDDEN + n0 + 16]);
      }
      Bf0[kc] = __builtin_bit_cast(v16bf, t0);
      Bf1[kc] = __builtin_bit_cast(v16bf, t1);
    }
  }
  // W3 B-fragment (zero-padded 32x16, only N<3 real). Wave w covers K slice [32w,32w+32).
  v16bf w3f;
  {
    v8u tw;
#pragma unroll
    for (int j = 0; j < 8; ++j) {
      const int k = wid * 32 + half * 16 + 2 * j;
      const float lo = (lm < SDIM) ? W3[k * SDIM + lm]       : 0.0f;
      const float hi = (lm < SDIM) ? W3[(k + 1) * SDIM + lm] : 0.0f;
      tw[j] = packbf(lo, hi);
    }
    w3f = __builtin_bit_cast(v16bf, tw);
  }

  // RK4 scalar lane bookkeeping (threads 0..47 manage the 16x3 state)
  const int rr = tid / 3;
  const int dd = tid - rr * 3;
  float b3r = 0.0f;
  if (tid < 48) {
    b3r = b3[dd];
    const float v = x0[(b0 + rr) * 3 + dd];
    sS[rr * 4 + dd]  = v;
    sSe[rr * 4 + dd] = v;
    sK[rr * 4 + dd]  = 0.0f;
    out[(size_t)(b0 + rr) * T * 3 + dd] = v;   // rollout[0] = initial state
  }

#pragma unroll 1
  for (int t = 0; t < T - 1; ++t) {
    const float dtv = tg[t + 1] - tg[t];       // uniform -> scalar load

#pragma unroll 1
    for (int stage = 0; stage < 4; ++stage) {
      __syncthreads();                         // (a) sSe ready; sOut of prev stage consumed

      // ---- layer 1: h1[r][tid] = tanh(sSe[r] . W1[:,tid] + b1[tid]) ----
#pragma unroll
      for (int r = 0; r < ROWS; ++r) {
        float a = fmaf(sSe[r * 4 + 0], w1r0, b1r);
        a = fmaf(sSe[r * 4 + 1], w1r1, a);
        a = fmaf(sSe[r * 4 + 2], w1r2, a);
        sH1[r * HSTR + tid] = f2bf(FAST_TANH(a));
      }
      if (tid < 64) sOut[tid] = 0.0f;
      __syncthreads();                         // (b) sH1 ready, sOut zeroed

      // ---- layer 2: [16x256] @ [256x256] via 16 WMMAs per wave ----
      v8f acc0 = {}, acc1 = {};
#pragma unroll
      for (int kc = 0; kc < 8; ++kc) {
        v8u av;
#pragma unroll
        for (int j = 0; j < 8; ++j) {
          // 16x32 bf16 A, contiguous k order (matches B): pair j -> K = half*16 + j*2
          const int kk = kc * 32 + half * 16 + 2 * j;
          av[j] = *(const unsigned int*)&sH1[lm * HSTR + kk];
        }
        const v16bf a = __builtin_bit_cast(v16bf, av);
        acc0 = __builtin_amdgcn_wmma_f32_16x16x32_bf16(false, a, false, Bf0[kc],
                                                       (short)0, acc0, false, false);
        acc1 = __builtin_amdgcn_wmma_f32_16x16x32_bf16(false, a, false, Bf1[kc],
                                                       (short)0, acc1, false, false);
      }

      // ---- tanh + bias, write h2 to LDS (C layout: VGPR v -> row v+8*half, col = N) ----
#pragma unroll
      for (int v = 0; v < 8; ++v) {
        const int m = v + half * 8;
        sH2[m * HSTR + wid * 32 + lm]      = f2bf(FAST_TANH(acc0[v] + b2r0));
        sH2[m * HSTR + wid * 32 + 16 + lm] = f2bf(FAST_TANH(acc1[v] + b2r1));
      }
      __syncthreads();                         // (c) sH2 ready

      // ---- layer 3: each wave does one partial 16x16x32 WMMA over its K slice ----
      v8u av2;
#pragma unroll
      for (int j = 0; j < 8; ++j) {
        const int kk = wid * 32 + half * 16 + 2 * j;
        av2[j] = *(const unsigned int*)&sH2[lm * HSTR + kk];
      }
      v8f accW = {};
      accW = __builtin_amdgcn_wmma_f32_16x16x32_bf16(false, __builtin_bit_cast(v16bf, av2),
                                                     false, w3f, (short)0, accW, false, false);
      if (lm < SDIM) {
#pragma unroll
        for (int v = 0; v < 8; ++v)
          atomicAdd(&sOut[(v + half * 8) * 4 + lm], accW[v]);  // ds_add_f32
      }
      __syncthreads();                         // (d) sOut complete

      // ---- RK4 bookkeeping ----
      if (tid < 48) {
        const float k = sOut[rr * 4 + dd] + b3r;
        const float wcoef = (stage == 0 || stage == 3) ? 1.0f : 2.0f;
        sK[rr * 4 + dd] += wcoef * k;
        if (stage < 3) {
          const float mc = (stage == 2) ? 1.0f : 0.5f;
          sSe[rr * 4 + dd] = sS[rr * 4 + dd] + mc * dtv * k;
        }
      }
    }

    if (tid < 48) {
      const float s = sS[rr * 4 + dd] + (dtv * (1.0f / 6.0f)) * sK[rr * 4 + dd];
      sS[rr * 4 + dd]  = s;
      sSe[rr * 4 + dd] = s;
      sK[rr * 4 + dd]  = 0.0f;
      out[(size_t)(b0 + rr) * T * 3 + (size_t)(t + 1) * 3 + dd] = s;
    }
  }
}

extern "C" void kernel_launch(void* const* d_in, const int* in_sizes, int n_in,
                              void* d_out, int out_size, void* d_ws, size_t ws_size,
                              hipStream_t stream) {
  const float* x0 = (const float*)d_in[0];
  const float* tg = (const float*)d_in[1];
  const float* W1 = (const float*)d_in[2];
  const float* b1 = (const float*)d_in[3];
  const float* W2 = (const float*)d_in[4];
  const float* b2 = (const float*)d_in[5];
  const float* W3 = (const float*)d_in[6];
  const float* b3 = (const float*)d_in[7];
  float* out = (float*)d_out;

  const int B = in_sizes[0] / SDIM;   // 8192
  const int T = in_sizes[1];          // 1000

  dim3 grid(B / ROWS);                // 512 workgroups, 16 trajectories each
  node_rk4_wmma<<<grid, 256, 0, stream>>>(x0, tg, W1, b1, W2, b2, W3, b3, out, T);
}
